// GAT_14894946583456
// MI455X (gfx1250) — matrix operand order repaired and checked by reading, not compiled, not run on hardware
//
#include <hip/hip_runtime.h>

#define N_NODES 4096
#define FEAT    256
#define HEADS   4
#define FPH     64           // features per head
#define KD      (HEADS*FPH)  // 256

typedef __attribute__((ext_vector_type(16))) __bf16 v16bf;
typedef __attribute__((ext_vector_type(8)))  __bf16 v8bf;
typedef __attribute__((ext_vector_type(8)))  float  v8f;
typedef __attribute__((ext_vector_type(4)))  int    v4i_;

#define AS_GLOBAL __attribute__((address_space(1)))
#define AS_LDS    __attribute__((address_space(3)))

// gfx1250 async global->LDS copy (ASYNCcnt path), if the toolchain exposes it
#if defined(__has_builtin)
# if __has_builtin(__builtin_amdgcn_global_load_async_to_lds_b128)
#  define HAVE_ASYNC_LDS 1
# endif
#endif

// ---------------------------------------------------------------------------
// CDNA5 WMMA fragment loaders (CDNA5 ISA §7.12.2, wave32)
// A 16x32 bf16: lanes 0-15 -> row M=lane, K {0..7} (v0-3) & {16..23} (v4-7)
//               lanes16-31 -> row M=lane-16, K {8..15} & {24..31}
// ---------------------------------------------------------------------------
__device__ __forceinline__ v16bf load_a_frag(const __bf16* As, int lane, int stride) {
  const int row  = lane & 15;
  const int half = lane >> 4;
  const __bf16* p0 = As + row * stride + half * 8;   // K 0..7  or 8..15
  const __bf16* p1 = p0 + 16;                        // K 16..23 or 24..31
  v8bf lo = *(const v8bf*)p0;
  v8bf hi = *(const v8bf*)p1;
  v16bf a;
#pragma unroll
  for (int i = 0; i < 8; ++i) { a[i] = lo[i]; a[i + 8] = hi[i]; }
  return a;
}

// B 32x16 bf16: lanes 0-15 hold K {0..7} & {16..23} of column N=lane,
//               lanes16-31 hold K {8..15} & {24..31} of column N=lane-16.
// BsT is stored transposed in LDS: BsT[col][K] -> two contiguous 8-element loads.
__device__ __forceinline__ v16bf load_b_frag(const __bf16* BsT, int lane,
                                             int colBase, int stride, int kOff) {
  const int col = colBase + (lane & 15);
  const int k0  = (lane >> 4) * 8;
  const __bf16* p = BsT + col * stride + kOff + k0;
  v8bf lo = *(const v8bf*)p;
  v8bf hi = *(const v8bf*)(p + 16);
  v16bf b;
#pragma unroll
  for (int i = 0; i < 8; ++i) { b[i] = lo[i]; b[i + 8] = hi[i]; }
  return b;
}

// ---------------------------------------------------------------------------
// Kernel 1: x_proj = x @ W + b  (bf16 WMMA, f32 accumulate)
// writes xp (f32, [n][j]) and xpT (bf16, [j][n] == [k][f][m]) for kernel 5.
// grid (N/64, KD/64), block 256 (8 waves); 64x64 tile, K-loop step 32.
// ---------------------------------------------------------------------------
__global__ __launch_bounds__(256)
void k_proj(const float* __restrict__ x, const float* __restrict__ W,
            const float* __restrict__ bias, float* __restrict__ xp,
            __bf16* __restrict__ xpT) {
  __shared__ __bf16 As[64 * 32];   // x tile   [row][kk]
  __shared__ __bf16 Bs[64 * 32];   // W tile transposed [j][kk]
  const int t = threadIdx.x;
  const int lane = t & 31, wave = t >> 5;
  const int n0 = blockIdx.x * 64;
  const int j0 = blockIdx.y * 64;
  const int ti  = wave & 3;          // output row-tile (16 rows)
  const int tj0 = (wave >> 2) * 2;   // two output col-tiles

  v8f c0 = {}; v8f c1 = {};

  for (int kk0 = 0; kk0 < FEAT; kk0 += 32) {
    { // stage x 64x32 -> As (f32 -> bf16)
      const int row = t >> 2, c8 = (t & 3) * 8;
      const float* src = x + (size_t)(n0 + row) * FEAT + kk0 + c8;
      float4 f0 = *(const float4*)src;
      float4 f1 = *(const float4*)(src + 4);
      __bf16* dst = As + row * 32 + c8;
      dst[0] = (__bf16)f0.x; dst[1] = (__bf16)f0.y;
      dst[2] = (__bf16)f0.z; dst[3] = (__bf16)f0.w;
      dst[4] = (__bf16)f1.x; dst[5] = (__bf16)f1.y;
      dst[6] = (__bf16)f1.z; dst[7] = (__bf16)f1.w;
    }
    { // stage W 32x64 -> Bs transposed [j][kk]
      const int kk = t >> 3, j8 = (t & 7) * 8;
      const float* src = W + (size_t)(kk0 + kk) * KD + j0 + j8;
      float4 f0 = *(const float4*)src;
      float4 f1 = *(const float4*)(src + 4);
      Bs[(j8 + 0) * 32 + kk] = (__bf16)f0.x;
      Bs[(j8 + 1) * 32 + kk] = (__bf16)f0.y;
      Bs[(j8 + 2) * 32 + kk] = (__bf16)f0.z;
      Bs[(j8 + 3) * 32 + kk] = (__bf16)f0.w;
      Bs[(j8 + 4) * 32 + kk] = (__bf16)f1.x;
      Bs[(j8 + 5) * 32 + kk] = (__bf16)f1.y;
      Bs[(j8 + 6) * 32 + kk] = (__bf16)f1.z;
      Bs[(j8 + 7) * 32 + kk] = (__bf16)f1.w;
    }
    __syncthreads();
    v16bf a  = load_a_frag(As + ti * 16 * 32, lane, 32);
    v16bf b0 = load_b_frag(Bs, lane, tj0 * 16, 32, 0);
    v16bf b1 = load_b_frag(Bs, lane, (tj0 + 1) * 16, 32, 0);
    c0 = __builtin_amdgcn_wmma_f32_16x16x32_bf16(false, a, false, b0, (short)0, c0, false, false);
    c1 = __builtin_amdgcn_wmma_f32_16x16x32_bf16(false, a, false, b1, (short)0, c1, false, false);
    __syncthreads();
  }

  // epilogue: +bias, write fp32 and transposed bf16 copies
  const int half = lane >> 4, nc = lane & 15;
#pragma unroll
  for (int r = 0; r < 8; ++r) {
    const int row = n0 + ti * 16 + r + 8 * half;   // D: VGPR r = M r / r+8
    {
      const int j = j0 + tj0 * 16 + nc;
      const float v = c0[r] + bias[j];
      xp[(size_t)row * KD + j] = v;
      xpT[(size_t)j * N_NODES + row] = (__bf16)v;
    }
    {
      const int j = j0 + (tj0 + 1) * 16 + nc;
      const float v = c1[r] + bias[j];
      xp[(size_t)row * KD + j] = v;
      xpT[(size_t)j * N_NODES + row] = (__bf16)v;
    }
  }
}

// ---------------------------------------------------------------------------
// Kernel 2: ss[n,k] = <xp[n,k,:], a_left[k,:]>, st likewise with a_right.
// One wave per node: 32 lanes x 8 floats = full 256-wide row; shfl reduce
// within each head's 8-lane group.
// ---------------------------------------------------------------------------
__global__ __launch_bounds__(256)
void k_scores(const float* __restrict__ xp, const float* __restrict__ al,
              const float* __restrict__ ar, float* __restrict__ ss,
              float* __restrict__ st) {
  const int wave = threadIdx.x >> 5, lane = threadIdx.x & 31;
  const int n = blockIdx.x * 8 + wave;
  const int k  = lane >> 3;
  const int f0 = (lane & 7) * 8;
  const float* xr  = xp + (size_t)n * KD + lane * 8;   // = k*FPH + f0
  const float* alp = al + k * FPH + f0;
  const float* arp = ar + k * FPH + f0;
  float pl = 0.f, pr = 0.f;
#pragma unroll
  for (int i = 0; i < 8; ++i) {
    const float v = xr[i];
    pl += v * alp[i];
    pr += v * arp[i];
  }
  pl += __shfl_xor(pl, 1); pl += __shfl_xor(pl, 2); pl += __shfl_xor(pl, 4);
  pr += __shfl_xor(pr, 1); pr += __shfl_xor(pr, 2); pr += __shfl_xor(pr, 4);
  if ((lane & 7) == 0) {
    ss[n * HEADS + k] = pl;
    st[n * HEADS + k] = pr;
  }
}

// ---------------------------------------------------------------------------
// Kernel 3: global sum / sumsq of z = lrelu(ss[n,k]+st[m,k]) over n,m,k.
// Never materializes the 256 MB score tensor. grid (256, HEADS).
// ---------------------------------------------------------------------------
__global__ __launch_bounds__(256)
void k_stats(const float* __restrict__ ss, const float* __restrict__ st,
             double* __restrict__ stats) {
  __shared__ float st_s[N_NODES];
  const int t  = threadIdx.x;
  const int k  = blockIdx.y;
  const int n0 = blockIdx.x * 16;
  for (int m = t; m < N_NODES; m += 256) st_s[m] = st[m * HEADS + k];
  __syncthreads();
  float s = 0.f, q = 0.f;
  for (int i = 0; i < 16; ++i) {
    const float ssv = ss[(n0 + i) * HEADS + k];
    for (int m = t; m < N_NODES; m += 256) {
      float z = ssv + st_s[m];
      z = z > 0.f ? z : 0.2f * z;
      s += z; q += z * z;
    }
  }
  __shared__ float rs[256], rq[256];
  rs[t] = s; rq[t] = q; __syncthreads();
  for (int w = 128; w > 0; w >>= 1) {
    if (t < w) { rs[t] += rs[t + w]; rq[t] += rq[t + w]; }
    __syncthreads();
  }
  if (t == 0) {
    atomicAdd(&stats[0], (double)rs[0]);
    atomicAdd(&stats[1], (double)rq[0]);
  }
}

// ---------------------------------------------------------------------------
// Kernel 4: invden[n,k] = 1 / sum_m exp((z-mu)*is) * mask[n,m]. One block per
// row; the one pass that must stream all 64 MB of mask from HBM (and leaves
// it hot in the 192 MB L2 for kernel 5).
// ---------------------------------------------------------------------------
__global__ __launch_bounds__(256)
void k_denom(const float* __restrict__ mask, const float* __restrict__ ss,
             const float* __restrict__ st, const double* __restrict__ stats,
             float* __restrict__ invden) {
  const int t = threadIdx.x;
  const int n = blockIdx.x;
  const double S = stats[0], Q = stats[1];
  const double cnt = (double)N_NODES * (double)N_NODES * (double)HEADS;
  const float mu = (float)(S / cnt);
  const float is = rsqrtf((float)((Q - S * S / cnt) / (cnt - 1.0)));
  const float4 ssv = *(const float4*)(ss + n * HEADS);
  float a0 = 0.f, a1 = 0.f, a2 = 0.f, a3 = 0.f;
  const float* mrow = mask + (size_t)n * N_NODES;
  for (int m = t; m < N_NODES; m += 256) {
    const float mv = mrow[m];
    if (mv != 0.f) {  // ~0.8% density: skip exp for the rest
      const float4 stv = *(const float4*)(st + m * HEADS);
      float z;
      z = ssv.x + stv.x; z = z > 0.f ? z : 0.2f * z; a0 += mv * __expf((z - mu) * is);
      z = ssv.y + stv.y; z = z > 0.f ? z : 0.2f * z; a1 += mv * __expf((z - mu) * is);
      z = ssv.z + stv.z; z = z > 0.f ? z : 0.2f * z; a2 += mv * __expf((z - mu) * is);
      z = ssv.w + stv.w; z = z > 0.f ? z : 0.2f * z; a3 += mv * __expf((z - mu) * is);
    }
  }
  __shared__ float4 red[256];
  red[t] = make_float4(a0, a1, a2, a3);
  __syncthreads();
  for (int w = 128; w > 0; w >>= 1) {
    if (t < w) {
      float4 o = red[t + w];
      red[t].x += o.x; red[t].y += o.y; red[t].z += o.z; red[t].w += o.w;
    }
    __syncthreads();
  }
  if (t == 0) {
    const float4 d = red[0];
    invden[n * HEADS + 0] = 1.f / d.x;
    invden[n * HEADS + 1] = 1.f / d.y;
    invden[n * HEADS + 2] = 1.f / d.z;
    invden[n * HEADS + 3] = 1.f / d.w;
  }
}

// ---------------------------------------------------------------------------
// Kernel 5: h[n,k,:] = ELU( sum_m aij[n,m,k] * x_proj[m,k,:] ).
// Attention tiles are regenerated on the fly (bf16) in LDS from the mask row
// (L2-resident) + ss/st/invden, then fed to WMMA against async-streamed xpT
// tiles. K-chunk of 64 -> 4 WMMAs per wave between barriers.
// grid (N/64, HEADS), block 256 = 8 waves; 64-row x 64-feat output per block.
// ---------------------------------------------------------------------------
__global__ __launch_bounds__(256)
void k_aggregate(const float* __restrict__ mask, const float* __restrict__ ss,
                 const float* __restrict__ st, const float* __restrict__ invden,
                 const double* __restrict__ stats, const __bf16* __restrict__ xpT,
                 float* __restrict__ out) {
  __shared__ __bf16 As[64 * 64];     // attention tile [row][m]   (8 KB)
  __shared__ __bf16 Bs[64 * 64];     // x_proj tile    [f][m]     (8 KB)
  __shared__ float  st_s[N_NODES];   // st for this head          (16 KB)
  const int t = threadIdx.x, lane = t & 31, wave = t >> 5;
  const int n0 = blockIdx.x * 64;
  const int k  = blockIdx.y;

  for (int m = t; m < N_NODES; m += 256) st_s[m] = st[m * HEADS + k];

  const double S = stats[0], Q = stats[1];
  const double cnt = (double)N_NODES * (double)N_NODES * (double)HEADS;
  const float mu = (float)(S / cnt);
  const float is = rsqrtf((float)((Q - S * S / cnt) / (cnt - 1.0)));

  // attention staging: thread t owns row ar, 8+8 columns at ac8 and ac8+32
  const int ar  = t >> 2;
  const int ac8 = (t & 3) * 8;
  const float ssn = ss[(n0 + ar) * HEADS + k];
  const float idn = invden[(n0 + ar) * HEADS + k];
  const float* mrow = mask + (size_t)(n0 + ar) * N_NODES;
  // xpT staging: thread t owns feature row br, 8+8 m's at bc8 and bc8+32
  const int br  = t >> 2;
  const int bc8 = (t & 3) * 8;
  const __bf16* xrow = xpT + (size_t)(k * FPH + br) * N_NODES;

  const int ti  = wave & 3;
  const int tj0 = (wave >> 2) * 2;
  v8f c0 = {}; v8f c1 = {};
  __syncthreads();

  for (int m0 = 0; m0 < N_NODES; m0 += 64) {
    if (m0 + 64 < N_NODES)
      __builtin_prefetch(mrow + m0 + 64 + ac8, 0, 0);   // global_prefetch_b8

    // x_proj tile: pure byte copy global -> LDS; use the gfx1250 async path.
    // Builtin signature (from clang diagnostic): arg0 = v4i in global AS,
    // arg1 = LDS destination, then imm offset + cpol.
#ifdef HAVE_ASYNC_LDS
    __builtin_amdgcn_global_load_async_to_lds_b128(
        (AS_GLOBAL v4i_*)(xrow + m0 + bc8),
        (AS_LDS    v4i_*)(Bs + br * 64 + bc8), 0, 0);
    __builtin_amdgcn_global_load_async_to_lds_b128(
        (AS_GLOBAL v4i_*)(xrow + m0 + 32 + bc8),
        (AS_LDS    v4i_*)(Bs + br * 64 + 32 + bc8), 0, 0);
#else
    *(v8bf*)(Bs + br * 64 + bc8)      = *(const v8bf*)(xrow + m0 + bc8);
    *(v8bf*)(Bs + br * 64 + 32 + bc8) = *(const v8bf*)(xrow + m0 + 32 + bc8);
#endif

    // build 64x64 bf16 attention tile (two 8-wide chunks per thread)
#pragma unroll
    for (int h = 0; h < 2; ++h) {
      const int cb = ac8 + h * 32;
      const float4 mv0 = *(const float4*)(mrow + m0 + cb);
      const float4 mv1 = *(const float4*)(mrow + m0 + cb + 4);
      const float mvs[8] = {mv0.x, mv0.y, mv0.z, mv0.w, mv1.x, mv1.y, mv1.z, mv1.w};
      __bf16* dst = As + ar * 64 + cb;
#pragma unroll
      for (int j = 0; j < 8; ++j) {
        float a = 0.f;
        const float mv = mvs[j];
        if (mv != 0.f) {
          float z = ssn + st_s[m0 + cb + j];
          z = z > 0.f ? z : 0.2f * z;
          a = mv * __expf((z - mu) * is) * idn;
        }
        dst[j] = (__bf16)a;
      }
    }

#ifdef HAVE_ASYNC_LDS
# if defined(__has_builtin) && __has_builtin(__builtin_amdgcn_s_wait_asynccnt)
    __builtin_amdgcn_s_wait_asynccnt(0);
# else
    asm volatile("s_wait_asynccnt 0" ::: "memory");
# endif
#endif
    __syncthreads();

#pragma unroll
    for (int c = 0; c < 2; ++c) {   // two K=32 WMMA steps per staged chunk
      v16bf a  = load_a_frag(As + ti * 16 * 64 + c * 32, lane, 64);
      v16bf b0 = load_b_frag(Bs, lane, tj0 * 16,       64, c * 32);
      v16bf b1 = load_b_frag(Bs, lane, (tj0 + 1) * 16, 64, c * 32);
      c0 = __builtin_amdgcn_wmma_f32_16x16x32_bf16(false, a, false, b0, (short)0, c0, false, false);
      c1 = __builtin_amdgcn_wmma_f32_16x16x32_bf16(false, a, false, b1, (short)0, c1, false, false);
    }
    __syncthreads();
  }

  // epilogue: ELU + store (out[n][k*64+f], f32)
  const int half = lane >> 4, nc = lane & 15;
#pragma unroll
  for (int r = 0; r < 8; ++r) {
    const int row = n0 + ti * 16 + r + 8 * half;
    {
      float v = c0[r];
      v = v > 0.f ? v : (__expf(v) - 1.f);
      out[(size_t)row * KD + k * FPH + tj0 * 16 + nc] = v;
    }
    {
      float v = c1[r];
      v = v > 0.f ? v : (__expf(v) - 1.f);
      out[(size_t)row * KD + k * FPH + (tj0 + 1) * 16 + nc] = v;
    }
  }
}

// ---------------------------------------------------------------------------
extern "C" void kernel_launch(void* const* d_in, const int* in_sizes, int n_in,
                              void* d_out, int out_size, void* d_ws, size_t ws_size,
                              hipStream_t stream) {
  const float* x    = (const float*)d_in[0];
  const float* mask = (const float*)d_in[1];
  // d_in[2] = batch (unused: single graph)
  const float* W    = (const float*)d_in[3];
  const float* b    = (const float*)d_in[4];
  const float* al   = (const float*)d_in[5];
  const float* ar   = (const float*)d_in[6];
  float* out = (float*)d_out;

  // workspace layout (~6.3 MB)
  char*   ws    = (char*)d_ws;
  double* stats = (double*)ws;                                     // 16 B (zeroed)
  float*  xp    = (float*)(ws + 256);                              // 4 MB
  __bf16* xpT   = (__bf16*)(ws + 256 + (size_t)N_NODES * KD * 4);  // 2 MB
  float*  ss    = (float*)((char*)xpT + (size_t)N_NODES * KD * 2); // 64 KB
  float*  st    = ss + N_NODES * HEADS;                            // 64 KB
  float*  invd  = st + N_NODES * HEADS;                            // 64 KB

  (void)hipMemsetAsync(stats, 0, 2 * sizeof(double), stream);
  k_proj<<<dim3(N_NODES / 64, KD / 64), 256, 0, stream>>>(x, W, b, xp, xpT);
  k_scores<<<N_NODES / 8, 256, 0, stream>>>(xp, al, ar, ss, st);
  k_stats<<<dim3(256, HEADS), 256, 0, stream>>>(ss, st, stats);
  k_denom<<<N_NODES, 256, 0, stream>>>(mask, ss, st, stats, invd);
  k_aggregate<<<dim3(N_NODES / 64, HEADS), 256, 0, stream>>>(mask, ss, st, invd, stats, xpT, out);
}
